// MultiHeadAttention_66116726555351
// MI455X (gfx1250) — matrix operand order repaired and checked by reading, not compiled, not run on hardware
//
#include <hip/hip_runtime.h>

// ---- shapes ----
#define BB 2
#define NN 2048
#define CC 768
#define HH 12
#define HD 64
#define ROWS (BB*NN)          // 4096
#define C2  (2*CC)            // 1536
#define CQKV (3*CC)           // 2304

#define AS1 __attribute__((address_space(1)))
#define AS3 __attribute__((address_space(3)))

#if defined(__HIP_DEVICE_COMPILE__) && __has_builtin(__builtin_amdgcn_global_load_async_to_lds_b128)
#define HAVE_ASYNC_LDS 1
#else
#define HAVE_ASYNC_LDS 0
#endif

typedef _Float16 v8h  __attribute__((ext_vector_type(8)));
typedef _Float16 v4h  __attribute__((ext_vector_type(4)));
typedef _Float16 v16h __attribute__((ext_vector_type(16)));
typedef float    v8f  __attribute__((ext_vector_type(8)));
typedef int      v4i  __attribute__((ext_vector_type(4)));

template <int N>
__device__ __forceinline__ void wait_asynccnt() {
#if HAVE_ASYNC_LDS
#if __has_builtin(__builtin_amdgcn_s_wait_asynccnt)
    __builtin_amdgcn_s_wait_asynccnt(N);
#else
    asm volatile("s_wait_asynccnt %0" ::"i"(N) : "memory");
#endif
#endif
}

// A fragment (16x32 f16, row-major): lane holds row r; elems 0..7 -> K=hi*8+0..7,
// elems 8..15 -> K=16+hi*8+0..7
__device__ __forceinline__ v16h ldA(const _Float16* rowp, int hi) {
    v8h lo = *(const v8h*)(rowp + hi * 8);
    v8h hh = *(const v8h*)(rowp + 16 + hi * 8);
    v16h a;
#pragma unroll
    for (int i = 0; i < 8; ++i) { a[i] = lo[i]; a[i + 8] = hh[i]; }
    return a;
}

// B fragment (32x16 f16) from TRANSPOSED source (BT[n][k]): lane holds column r;
// elems 0..7 -> K=hi*8+0..7, elems 8..15 -> K=16+hi*8+0..7
__device__ __forceinline__ v16h ldB(const _Float16* colp, int hi) {
    v8h x0 = *(const v8h*)(colp + hi * 8);
    v8h x1 = *(const v8h*)(colp + 16 + hi * 8);
    v16h b;
#pragma unroll
    for (int i = 0; i < 8; ++i) { b[i] = x0[i]; b[i + 8] = x1[i]; }
    return b;
}

__device__ __forceinline__ v8f wmma32(v16h a, v16h b, v8f c) {
    return __builtin_amdgcn_wmma_f32_16x16x32_f16(false, a, false, b,
                                                  (short)0, c, false, false);
}

// ---------------- conversion kernels ----------------
__global__ void k_f32_to_f16(const float4* __restrict__ src,
                             _Float16* __restrict__ dst, int n4) {
    int i = blockIdx.x * blockDim.x + threadIdx.x;
    if (i < n4) {
        float4 v = src[i];
        v4h o = { (_Float16)v.x, (_Float16)v.y, (_Float16)v.z, (_Float16)v.w };
        *(v4h*)(dst + (size_t)i * 4) = o;
    }
}

__global__ void k_build_wt(const float* __restrict__ Wq,
                           const float* __restrict__ Wkv,
                           _Float16* __restrict__ WT) {
    int i = blockIdx.x * blockDim.x + threadIdx.x;
    if (i >= CQKV * CC) return;
    int j = i / CC, k = i % CC;
    float v = (j < CC) ? Wq[(size_t)k * CC + j] : Wkv[(size_t)k * C2 + (j - CC)];
    WT[i] = (_Float16)v;
}

__global__ void k_build_wpt(const float* __restrict__ Wp,
                            _Float16* __restrict__ WT) {
    int i = blockIdx.x * blockDim.x + threadIdx.x;
    if (i >= CC * CC) return;
    int j = i / CC, k = i % CC;
    WT[i] = (_Float16)Wp[(size_t)k * CC + j];
}

// Shared GEMM core: 32x64 tile, K=CC, ping-pong register banks (no swap moves).
template <typename EPI>
__device__ __forceinline__ void gemm32x64(const _Float16* ar0, const _Float16* ar1,
                                          const _Float16* w0, int hi, int r,
                                          EPI epi) {
    v8f acc[2][4] = {};
    v16h a0A = ldA(ar0, hi), a1A = ldA(ar1, hi);
    v16h bA[4], a0B, a1B, bB[4];
#pragma unroll
    for (int t = 0; t < 4; ++t) bA[t] = ldB(w0 + (size_t)t * 16 * CC, hi);

    for (int k0 = 0; k0 < CC - 64; k0 += 64) {
        a0B = ldA(ar0 + k0 + 32, hi);
        a1B = ldA(ar1 + k0 + 32, hi);
#pragma unroll
        for (int t = 0; t < 4; ++t) bB[t] = ldB(w0 + (size_t)t * 16 * CC + k0 + 32, hi);
#pragma unroll
        for (int t = 0; t < 4; ++t) {
            acc[0][t] = wmma32(a0A, bA[t], acc[0][t]);
            acc[1][t] = wmma32(a1A, bA[t], acc[1][t]);
        }
        a0A = ldA(ar0 + k0 + 64, hi);
        a1A = ldA(ar1 + k0 + 64, hi);
#pragma unroll
        for (int t = 0; t < 4; ++t) bB[t] = bB[t], bA[t] = ldB(w0 + (size_t)t * 16 * CC + k0 + 64, hi);
#pragma unroll
        for (int t = 0; t < 4; ++t) {
            acc[0][t] = wmma32(a0B, bB[t], acc[0][t]);
            acc[1][t] = wmma32(a1B, bB[t], acc[1][t]);
        }
    }
    // tail pair: bank A holds k=CC-64; load bank B at k=CC-32
    a0B = ldA(ar0 + CC - 32, hi);
    a1B = ldA(ar1 + CC - 32, hi);
#pragma unroll
    for (int t = 0; t < 4; ++t) bB[t] = ldB(w0 + (size_t)t * 16 * CC + CC - 32, hi);
#pragma unroll
    for (int t = 0; t < 4; ++t) {
        acc[0][t] = wmma32(a0A, bA[t], acc[0][t]);
        acc[1][t] = wmma32(a1A, bA[t], acc[1][t]);
    }
#pragma unroll
    for (int t = 0; t < 4; ++t) {
        acc[0][t] = wmma32(a0B, bB[t], acc[0][t]);
        acc[1][t] = wmma32(a1B, bB[t], acc[1][t]);
    }
    epi(acc);
}

// ---------------- QKV projection ----------------
// grid (128, 9), block 128: wave computes a 32x64 tile of (4096 x 2304)
__global__ __launch_bounds__(128) void k_qkv(const _Float16* __restrict__ X,
                                             const _Float16* __restrict__ WT,
                                             const float* __restrict__ bq,
                                             const float* __restrict__ bkv,
                                             _Float16* __restrict__ Q,
                                             _Float16* __restrict__ K,
                                             _Float16* __restrict__ VT) {
    int wave = threadIdx.x >> 5, lane = threadIdx.x & 31;
    int r = lane & 15, hi = lane >> 4;
    int row0 = blockIdx.x * 32;
    int col0 = (blockIdx.y * 4 + wave) * 64;
    const _Float16* ar0 = X + (size_t)(row0 + r) * CC;
    const _Float16* ar1 = X + (size_t)(row0 + 16 + r) * CC;
    const _Float16* w0  = WT + (size_t)(col0 + r) * CC;

    gemm32x64(ar0, ar1, w0, hi, r, [&](v8f acc[2][4]) {
#pragma unroll
        for (int g = 0; g < 2; ++g) {
#pragma unroll
            for (int t = 0; t < 4; ++t) {
                int col = col0 + t * 16 + r;
                float bias = (col < CC) ? bq[col] : bkv[col - CC];
#pragma unroll
                for (int e = 0; e < 8; ++e) {
                    int grow = row0 + g * 16 + hi * 8 + e;   // = b*N + n
                    int b = grow >> 11, n = grow & (NN - 1);
                    _Float16 hv = (_Float16)(acc[g][t][e] + bias);
                    if (col < CC) {
                        int h = col >> 6, d = col & 63;
                        Q[(((size_t)(b * HH + h)) * NN + n) * HD + d] = hv;
                    } else if (col < C2) {
                        int c2 = col - CC; int h = c2 >> 6, d = c2 & 63;
                        K[(((size_t)(b * HH + h)) * NN + n) * HD + d] = hv;
                    } else {
                        int c2 = col - C2; int h = c2 >> 6, d = c2 & 63;
                        VT[(((size_t)(b * HH + h)) * HD + d) * NN + n] = hv;
                    }
                }
            }
        }
    });
}

// ---------------- attention ----------------
// grid 768, block 128: each wave owns one (b, h, 16-query block)
__global__ __launch_bounds__(128) void k_attn(const _Float16* __restrict__ Q,
                                              const _Float16* __restrict__ K,
                                              const _Float16* __restrict__ VT,
                                              const unsigned char* __restrict__ mask,
                                              float* __restrict__ att,
                                              _Float16* __restrict__ CTX) {
    __shared__ float lM[4][16];
    __shared__ float lS[4][16];
#if HAVE_ASYNC_LDS
    __shared__ _Float16 kbuf[4][2][16 * HD];   // per-wave double-buffered K tile
#endif
    int wave = threadIdx.x >> 5, lane = threadIdx.x & 31;
    int r = lane & 15, hi = lane >> 4;
    int id = blockIdx.x * 4 + wave;
    int qb = id & 127, bh = id >> 7;
    int h = bh % HH, b = bh / HH;
    int n0 = qb * 16;
    const _Float16* Qp = Q + (size_t)bh * NN * HD;
    const _Float16* Kp = K + (size_t)bh * NN * HD;
    const _Float16* Vp = VT + (size_t)bh * HD * NN;
    const unsigned char* mk = mask + b * NN;
    const float scale = 0.125f;

    const _Float16* qrow = Qp + (size_t)(n0 + r) * HD;
    v16h a0 = ldA(qrow, hi), a1 = ldA(qrow + 32, hi);

    float mx[8], sm[8];
#pragma unroll
    for (int e = 0; e < 8; ++e) { mx[e] = -INFINITY; sm[e] = 0.f; }

    // scale+mask, store logits, update online (max, sum-exp)
    auto stats_store = [&](v8f c, int mt) {
        int mcol = mt * 16 + r;
        bool ok = mk[mcol] != 0;
        size_t abase = (((size_t)(b * NN + n0 + hi * 8)) * NN + mcol) * HH + h;
#pragma unroll
        for (int e = 0; e < 8; ++e) {
            float s = c[e] * scale, o;
            if (ok) {
                o = s;
                float nm = fmaxf(mx[e], s);
                sm[e] = sm[e] * __expf(mx[e] - nm) + __expf(s - nm);
                mx[e] = nm;
            } else {
                o = -INFINITY;
            }
            att[abase + (size_t)e * NN * HH] = o;
        }
    };

#if HAVE_ASYNC_LDS
    // ---- pass 1: async-copy K tiles into LDS ring, WMMA from LDS ----
    auto copy_tile = [&](int mt, int bufi) {
        char* g = (char*)(const char*)(Kp + (size_t)mt * 16 * HD);
        char* l = (char*)&kbuf[wave][bufi][0];
#pragma unroll
        for (int i = 0; i < 4; ++i) {
            int off = lane * 16 + i * 512;
            __builtin_amdgcn_global_load_async_to_lds_b128(
                (AS1 v4i*)(g + off), (AS3 v4i*)(l + off), 0, 0);
        }
    };
    auto proc_tile = [&](int mt, int bufi) {
        const _Float16* kt = &kbuf[wave][bufi][(size_t)r * HD];
        v8f c = {};
        c = wmma32(a0, ldB(kt, hi), c);
        c = wmma32(a1, ldB(kt + 32, hi), c);
        stats_store(c, mt);
    };
    copy_tile(0, 0);
    int mt = 0;
    for (; mt + 2 < 128; mt += 2) {       // unrolled x2: buffer index is constant
        copy_tile(mt + 1, 1);
        wait_asynccnt<4>();
        proc_tile(mt, 0);
        copy_tile(mt + 2, 0);
        wait_asynccnt<4>();
        proc_tile(mt + 1, 1);
    }
    // mt == 126 here: tile 126 already copied into buf 0
    copy_tile(127, 1);
    wait_asynccnt<4>();
    proc_tile(126, 0);
    wait_asynccnt<0>();
    proc_tile(127, 1);
#else
    // ---- pass 1 fallback: register ping-pong global loads ----
    v16h b0A = ldB(Kp + (size_t)r * HD, hi);
    v16h b1A = ldB(Kp + (size_t)r * HD + 32, hi);
    v16h b0B, b1B;
    for (int t2 = 0; t2 + 2 < 128; t2 += 2) {
        const _Float16* k1 = Kp + (size_t)((t2 + 1) * 16 + r) * HD;
        b0B = ldB(k1, hi); b1B = ldB(k1 + 32, hi);
        { v8f c = {}; c = wmma32(a0, b0A, c); c = wmma32(a1, b1A, c); stats_store(c, t2); }
        const _Float16* k2 = Kp + (size_t)((t2 + 2) * 16 + r) * HD;
        b0A = ldB(k2, hi); b1A = ldB(k2 + 32, hi);
        { v8f c = {}; c = wmma32(a0, b0B, c); c = wmma32(a1, b1B, c); stats_store(c, t2 + 1); }
    }
    {
        const _Float16* k1 = Kp + (size_t)(127 * 16 + r) * HD;
        b0B = ldB(k1, hi); b1B = ldB(k1 + 32, hi);
        { v8f c = {}; c = wmma32(a0, b0A, c); c = wmma32(a1, b1A, c); stats_store(c, 126); }
        { v8f c = {}; c = wmma32(a0, b0B, c); c = wmma32(a1, b1B, c); stats_store(c, 127); }
    }
#endif

    // merge (max,sum) across the 16 lanes holding one row
#pragma unroll
    for (int off = 1; off < 16; off <<= 1) {
#pragma unroll
        for (int e = 0; e < 8; ++e) {
            float om = __shfl_xor(mx[e], off, 32);
            float os = __shfl_xor(sm[e], off, 32);
            float nm = fmaxf(mx[e], om);
            float ns = (nm == -INFINITY)
                           ? 0.f
                           : sm[e] * __expf(mx[e] - nm) + os * __expf(om - nm);
            mx[e] = nm; sm[e] = ns;
        }
    }
    if (r == 0) {
#pragma unroll
        for (int e = 0; e < 8; ++e) {
            lM[wave][hi * 8 + e] = mx[e];
            lS[wave][hi * 8 + e] = sm[e];
        }
    }
    __syncthreads();
    float rM = lM[wave][r];
    float rS = lS[wave][r];
    float rI = (rS > 0.f) ? 1.0f / rS : 0.f;

    // ---- pass 2: normalize att (final output) + O = P V, ping-pong pipelined ----
    v8f acc[4] = {};
    int hi8 = hi * 8;
    size_t prowbase = ((size_t)(b * NN + n0 + r)) * NN * HH + h;

    auto gatherP = [&](int m0) -> v16h {
        v16h pa;
#pragma unroll
        for (int e = 0; e < 16; ++e) {
            int mo = (e < 8) ? (hi8 + e) : (16 + hi8 + (e - 8));
            size_t ai = prowbase + (size_t)(m0 + mo) * HH;
            float s = att[ai];
            float p = (s == -INFINITY) ? 0.f : __expf(s - rM) * rI;
            att[ai] = p;                   // final softmaxed attention
            pa[e] = (_Float16)p;
        }
        if (m0 + 128 < NN)
            __builtin_prefetch((const void*)(att + prowbase + (size_t)(m0 + 128) * HH), 0, 1);
        return pa;
    };
    auto vload = [&](v16h vb[4], int m0) {
#pragma unroll
        for (int t = 0; t < 4; ++t)
            vb[t] = ldB(Vp + (size_t)(t * 16 + r) * NN + m0, hi);
    };

    v16h paA = gatherP(0), paB;
    v16h vbA[4], vbB[4];
    vload(vbA, 0);
    for (int i = 0; i < 31; ++i) {        // steps 2i and 2i+1 of 64
        int m0 = i * 64;
        vload(vbB, m0 + 32);
        paB = gatherP(m0 + 32);
#pragma unroll
        for (int t = 0; t < 4; ++t) acc[t] = wmma32(paA, vbA[t], acc[t]);
        vload(vbA, m0 + 64);
        paA = gatherP(m0 + 64);
#pragma unroll
        for (int t = 0; t < 4; ++t) acc[t] = wmma32(paB, vbB[t], acc[t]);
    }
    vload(vbB, 2016);
    paB = gatherP(2016);
#pragma unroll
    for (int t = 0; t < 4; ++t) acc[t] = wmma32(paA, vbA[t], acc[t]);
#pragma unroll
    for (int t = 0; t < 4; ++t) acc[t] = wmma32(paB, vbB[t], acc[t]);

#pragma unroll
    for (int t = 0; t < 4; ++t) {
#pragma unroll
        for (int e = 0; e < 8; ++e) {
            int n = n0 + hi * 8 + e;
            CTX[((size_t)(b * NN + n)) * CC + h * HD + t * 16 + r] =
                (_Float16)acc[t][e];
        }
    }
}

// ---------------- output projection ----------------
// grid (128, 3), block 128: wave computes 32x64 tile of (4096 x 768)
__global__ __launch_bounds__(128) void k_proj(const _Float16* __restrict__ CTXh,
                                              const _Float16* __restrict__ WpT,
                                              const float* __restrict__ bp,
                                              float* __restrict__ OUT) {
    int wave = threadIdx.x >> 5, lane = threadIdx.x & 31;
    int r = lane & 15, hi = lane >> 4;
    int row0 = blockIdx.x * 32;
    int col0 = (blockIdx.y * 4 + wave) * 64;
    const _Float16* ar0 = CTXh + (size_t)(row0 + r) * CC;
    const _Float16* ar1 = CTXh + (size_t)(row0 + 16 + r) * CC;
    const _Float16* w0  = WpT + (size_t)(col0 + r) * CC;

    gemm32x64(ar0, ar1, w0, hi, r, [&](v8f acc[2][4]) {
#pragma unroll
        for (int g = 0; g < 2; ++g) {
#pragma unroll
            for (int t = 0; t < 4; ++t) {
                int col = col0 + t * 16 + r;
                float bias = bp[col];
#pragma unroll
                for (int e = 0; e < 8; ++e) {
                    int grow = row0 + g * 16 + hi * 8 + e;
                    OUT[(size_t)grow * CC + col] = acc[g][t][e] + bias;
                }
            }
        }
    });
}

extern "C" void kernel_launch(void* const* d_in, const int* in_sizes, int n_in,
                              void* d_out, int out_size, void* d_ws, size_t ws_size,
                              hipStream_t stream) {
    const float* x            = (const float*)d_in[0];
    const unsigned char* mask = (const unsigned char*)d_in[1]; // jax bool -> 1 byte
    const float* Wq           = (const float*)d_in[2];
    const float* bq           = (const float*)d_in[3];
    const float* Wkv          = (const float*)d_in[4];
    const float* bkv          = (const float*)d_in[5];
    const float* Wp           = (const float*)d_in[6];
    const float* bp           = (const float*)d_in[7];

    float* out = (float*)d_out;
    float* att = out + (size_t)ROWS * CC;

    _Float16* ws = (_Float16*)d_ws;
    size_t o = 0;
    _Float16* Xh  = ws + o; o += (size_t)ROWS * CC;
    _Float16* WT  = ws + o; o += (size_t)CQKV * CC;
    _Float16* WpT = ws + o; o += (size_t)CC * CC;
    _Float16* Qh  = ws + o; o += (size_t)BB * HH * NN * HD;
    _Float16* Kh  = ws + o; o += (size_t)BB * HH * NN * HD;
    _Float16* VTh = ws + o; o += (size_t)BB * HH * HD * NN;
    _Float16* CTX = ws + o; o += (size_t)ROWS * CC;

    int n4 = ROWS * CC / 4;
    k_f32_to_f16<<<(n4 + 255) / 256, 256, 0, stream>>>((const float4*)x, Xh, n4);
    k_build_wt<<<(CQKV * CC + 255) / 256, 256, 0, stream>>>(Wq, Wkv, WT);
    k_build_wpt<<<(CC * CC + 255) / 256, 256, 0, stream>>>(Wp, WpT);

    k_qkv<<<dim3(ROWS / 32, 9), 128, 0, stream>>>(Xh, WT, bq, bkv, Qh, Kh, VTh);
    k_attn<<<(BB * HH * (NN / 16)) / 4, 128, 0, stream>>>(Qh, Kh, VTh, mask, att, CTX);
    k_proj<<<dim3(ROWS / 32, 3), 128, 0, stream>>>(CTX, WpT, bp, out);
}
